// PGCRN_23888608101394
// MI455X (gfx1250) — matrix operand order repaired and checked by reading, not compiled, not run on hardware
//
#include <hip/hip_runtime.h>
#include <math.h>

// ---------------------------------------------------------------------------
// PGCRN (Chebyshev graph-conv GRU) for MI455X / gfx1250.
// wave32 + v_wmma_f32_16x16x32_bf16 + TDM (tensor_load_to_lds) double-buffered
// tile staging with s_wait_tensorcnt.
// Shapes: B=8, T=12, N=2048, D=64, K_cheb=3 (T0=I handled implicitly).
// ---------------------------------------------------------------------------

typedef __bf16 bf16_t;
typedef __attribute__((ext_vector_type(16))) __bf16 v16bf;
typedef __attribute__((ext_vector_type(8)))  __bf16 v8bf;
typedef __attribute__((ext_vector_type(8)))  float  v8f;
typedef __attribute__((ext_vector_type(4)))  unsigned int v4u;
typedef __attribute__((ext_vector_type(8)))  int v8i;
typedef __attribute__((ext_vector_type(4)))  int v4i;

#define BK 32            // K-depth of one WMMA == LDS K tile
#define NN 2048          // nodes
#define DD 64            // features
#define BB 8             // batch
#define TT 12            // timesteps
#define KC 384           // K*2D = 3*128 (projection depth)

// ---------------------------------------------------------------------------
// TDM: DMA one 2D bf16 tile [tile_rows x BK] from row-major global (leading
// dim = ld elements) into LDS at byte offset lds_addr.  8-byte element mode:
// row = 64B = 8 elems.  Rows pack contiguously in LDS (row-major [rows][BK]).
// Issued per-wave (EXEC ignored); caller restricts to one wave per block.
// ---------------------------------------------------------------------------
__device__ __forceinline__ void tdm_load_tile(
    unsigned lds_addr, const bf16_t* gptr, unsigned tile_rows, long ld_elems)
{
    unsigned long long ga = (unsigned long long)gptr;
    unsigned long long s0 = ((unsigned long long)ld_elems * 2ull) >> 3; // 8B units
    unsigned td0   = (unsigned)s0;        // tensor_dim0 (>= tile0, no OOB)
    unsigned td1   = tile_rows;           // tensor_dim1
    unsigned tile0 = (BK * 2) >> 3;       // 8

    v4u g0;
    g0.x = 1u;                                            // count=1 (valid D#)
    g0.y = lds_addr;                                      // lds_addr
    g0.z = (unsigned)ga;                                  // global_addr[31:0]
    g0.w = (unsigned)((ga >> 32) & 0x01ffffffu) | (2u << 30); // ga[56:32]|type=2

    v8i g1;
    g1[0] = (int)(3u << 16);                              // data_size=8B
    g1[1] = (int)((td0 & 0xffffu) << 16);                 // tensor_dim0 lo
    g1[2] = (int)((td0 >> 16) | ((td1 & 0xffffu) << 16)); // td0 hi | td1 lo
    g1[3] = (int)((td1 >> 16) | (tile0 << 16));           // td1 hi | tile_dim0
    g1[4] = (int)(tile_rows & 0xffffu);                   // tile_dim1 (dim2=0)
    g1[5] = (int)(unsigned)(s0 & 0xffffffffull);          // dim0_stride lo
    g1[6] = (int)(unsigned)((s0 >> 32) & 0xffffull);      // dim0_stride hi
    g1[7] = 0;                                            // dim1_stride (unused)

    v4i g2 = {0, 0, 0, 0};
    v4i g3 = {0, 0, 0, 0};
#if __has_include(<hip/amd_detail/amd_gfx1250_TDM.h>)
    v8i g4 = {0, 0, 0, 0, 0, 0, 0, 0};
    __builtin_amdgcn_tensor_load_to_lds(g0, g1, g2, g3, g4, 0);
#else
    __builtin_amdgcn_tensor_load_to_lds(g0, g1, g2, g3, 0);
#endif
}

// ---------------------------------------------------------------------------
// Templated GEMM core.  C[BM x BN] = A[BM x K] * B[K x BN], bf16 in, f32 acc.
//   A  : row-major [M][K], lda in elements
//   Bt : B transposed, row-major [N][K], ldb in elements
// 256 threads = 8 waves (WAVES_M x WAVES_N); wave owns WM_T x WN_T 16x16 tiles.
// Double-buffered LDS; wave 0 drives the TDM pipeline.
// ---------------------------------------------------------------------------
template<int WAVES_M, int WAVES_N, int WM_T, int WN_T>
__device__ __forceinline__ void gemm_core(
    const bf16_t* __restrict__ A,  int lda,
    const bf16_t* __restrict__ Bt, int ldb,
    int K, int mblk, int nblk,
    v8f (&acc)[WM_T][WN_T])
{
    constexpr int BM = WAVES_M * WM_T * 16;
    constexpr int BN = WAVES_N * WN_T * 16;

    __shared__ bf16_t As[2][BM][BK];
    __shared__ bf16_t Bs[2][BN][BK];   // stored [n][k] (transposed B)

    const int tid  = threadIdx.x;
    const int lane = tid & 31;
    const int wave = tid >> 5;
    const int wm   = wave % WAVES_M;
    const int wn   = wave / WAVES_M;
    const int g    = lane >> 4;       // half-wave select
    const int l    = lane & 15;
    const long m0  = (long)mblk * BM;
    const long n0  = (long)nblk * BN;

    const unsigned as_base = (unsigned)(unsigned long long)&As[0][0][0];
    const unsigned bs_base = (unsigned)(unsigned long long)&Bs[0][0][0];
    constexpr unsigned ASZ = (unsigned)BM * BK * 2u;
    constexpr unsigned BSZ = (unsigned)BN * BK * 2u;

#pragma unroll
    for (int i = 0; i < WM_T; ++i)
#pragma unroll
        for (int j = 0; j < WN_T; ++j)
#pragma unroll
            for (int e = 0; e < 8; ++e) acc[i][j][e] = 0.f;

    if (wave == 0) {                        // prologue: stage buffer 0
        tdm_load_tile(as_base, A  + m0 * (long)lda, BM, lda);
        tdm_load_tile(bs_base, Bt + n0 * (long)ldb, BN, ldb);
    }

    int cur = 0;
    for (int k0 = 0; k0 < K; k0 += BK, cur ^= 1) {
        if (wave == 0)                      // DMA of buf[cur] complete
            __builtin_amdgcn_s_wait_tensorcnt((short)0);
        __syncthreads();                    // buf[cur] visible to all waves
        if (wave == 0 && k0 + BK < K) {     // prefetch next tile into buf[cur^1]
            tdm_load_tile(as_base + (unsigned)(cur ^ 1) * ASZ,
                          A  + m0 * (long)lda + (k0 + BK), BM, lda);
            tdm_load_tile(bs_base + (unsigned)(cur ^ 1) * BSZ,
                          Bt + n0 * (long)ldb + (k0 + BK), BN, ldb);
        }

        // ---- fragments per ISA VGPR layouts ----
        // A 16x32 bf16: lane(g,l): v[j] = A[m=l][(j>>3)*16 + g*8 + (j&7)]
        v16bf af[WM_T];
#pragma unroll
        for (int i = 0; i < WM_T; ++i) {
            int row = (wm * WM_T + i) * 16 + l;
            v8bf lo = *(const v8bf*)&As[cur][row][g * 8];
            v8bf hi = *(const v8bf*)&As[cur][row][16 + g * 8];
#pragma unroll
            for (int e = 0; e < 8; ++e) { af[i][e] = lo[e]; af[i][8 + e] = hi[e]; }
        }
        // B 32x16 bf16: lane(g,l): v[j] = B[k=(j>>3)*16 + g*8 + (j&7)][n=l]  (Bs is [n][k])
        v16bf bfv[WN_T];
#pragma unroll
        for (int j = 0; j < WN_T; ++j) {
            int row = (wn * WN_T + j) * 16 + l;
            v8bf lo = *(const v8bf*)&Bs[cur][row][g * 8];
            v8bf hi = *(const v8bf*)&Bs[cur][row][16 + g * 8];
#pragma unroll
            for (int e = 0; e < 8; ++e) { bfv[j][e] = lo[e]; bfv[j][8 + e] = hi[e]; }
        }

#pragma unroll
        for (int i = 0; i < WM_T; ++i)
#pragma unroll
            for (int j = 0; j < WN_T; ++j)
                acc[i][j] = __builtin_amdgcn_wmma_f32_16x16x32_bf16(
                    false, af[i], false, bfv[j], (short)0, acc[i][j],
                    false, false);
        // single barrier per iteration (top of loop) is sufficient:
        // LDS reads retire before each wave's WMMAs issue, which precede the
        // next barrier, which precedes the TDM that overwrites that buffer.
    }
}

// D layout: element (i,j,r) -> m = tile_m + 8*g + r, n = tile_n + l.

// ---------------------------------------------------------------------------
// Support precompute: A2 = 2*A@A - I   (bf16 out)
// ---------------------------------------------------------------------------
__global__ __launch_bounds__(256) void k_gemm_A2(
    const bf16_t* __restrict__ Abf, const bf16_t* __restrict__ AbfT,
    bf16_t* __restrict__ A2)
{
    v8f acc[4][2];
    gemm_core<2, 4, 4, 2>(Abf, NN, AbfT, NN, NN, blockIdx.y, blockIdx.x, acc);
    const int lane = threadIdx.x & 31, wave = threadIdx.x >> 5;
    const int wm = wave & 1, wn = wave >> 1, g = lane >> 4, l = lane & 15;
    const long m0 = (long)blockIdx.y * 128, n0 = (long)blockIdx.x * 128;
#pragma unroll
    for (int i = 0; i < 4; ++i)
#pragma unroll
        for (int j = 0; j < 2; ++j)
#pragma unroll
            for (int r = 0; r < 8; ++r) {
                long m = m0 + (wm * 4 + i) * 16 + g * 8 + r;
                long n = n0 + (wn * 2 + j) * 16 + l;
                float v = 2.f * acc[i][j][r] - (m == n ? 1.f : 0.f);
                A2[m * NN + n] = (bf16_t)v;
            }
}

// ---------------------------------------------------------------------------
// Spatial graph conv: Y_k = S_k @ Xpack   (S_k in {A, A2}, z = blockIdx.z)
//   XpT : [B*128][N] bf16 (Xpack transposed -> B operand)
//   Xg  : [B][N][384] bf16, slab k = z+1 written here
// ---------------------------------------------------------------------------
__global__ __launch_bounds__(256) void k_spatial(
    const bf16_t* __restrict__ A1, const bf16_t* __restrict__ A2,
    const bf16_t* __restrict__ XpT, bf16_t* __restrict__ Xg)
{
    const bf16_t* A = blockIdx.z ? A2 : A1;
    v8f acc[4][2];
    gemm_core<2, 4, 4, 2>(A, NN, XpT, NN, NN, blockIdx.y, blockIdx.x, acc);
    const int lane = threadIdx.x & 31, wave = threadIdx.x >> 5;
    const int wm = wave & 1, wn = wave >> 1, g = lane >> 4, l = lane & 15;
    const int kslab = blockIdx.z + 1;
    const long m0 = (long)blockIdx.y * 128, n0 = (long)blockIdx.x * 128;
#pragma unroll
    for (int i = 0; i < 4; ++i)
#pragma unroll
        for (int j = 0; j < 2; ++j)
#pragma unroll
            for (int r = 0; r < 8; ++r) {
                long node = m0 + (wm * 4 + i) * 16 + g * 8 + r;
                long col  = n0 + (wn * 2 + j) * 16 + l;     // 0..1023 = b*128+c
                long b = col >> 7, c = col & 127;
                Xg[(b * NN + node) * KC + kslab * 128 + c] = (bf16_t)acc[i][j][r];
            }
}

// ---------------------------------------------------------------------------
// Gate projection: ZR = sigmoid(Xg[b] @ Wg + bg); fuse z*h repack + r store
// ---------------------------------------------------------------------------
__global__ __launch_bounds__(256) void k_proj_gate(
    bf16_t* __restrict__ Xg, const bf16_t* __restrict__ WgT,
    const float* __restrict__ bg, const float* __restrict__ Hbuf,
    bf16_t* __restrict__ XpT, float* __restrict__ Rbuf)
{
    const int b = blockIdx.z;
    v8f acc[4][2];
    gemm_core<2, 4, 4, 2>(Xg + (long)b * NN * KC, KC, WgT, KC, KC,
                          blockIdx.y, 0, acc);
    const int lane = threadIdx.x & 31, wave = threadIdx.x >> 5;
    const int wm = wave & 1, wn = wave >> 1, g = lane >> 4, l = lane & 15;
    const long m0 = (long)blockIdx.y * 128;
#pragma unroll
    for (int i = 0; i < 4; ++i)
#pragma unroll
        for (int j = 0; j < 2; ++j)
#pragma unroll
            for (int r = 0; r < 8; ++r) {
                long node = m0 + (wm * 4 + i) * 16 + g * 8 + r;
                int  o    = (wn * 2 + j) * 16 + l;          // 0..127
                float s = 1.f / (1.f + expf(-(acc[i][j][r] + bg[o])));
                if (o < DD) {                               // z half -> z*h
                    float zh = s * Hbuf[((long)b * NN + node) * DD + o];
                    bf16_t v = (bf16_t)zh;
                    XpT[((long)(b * 128 + DD + o)) * NN + node] = v;
                    Xg [((long)b * NN + node) * KC + DD + o]    = v;
                } else {                                    // r half
                    Rbuf[((long)b * NN + node) * DD + (o - DD)] = s;
                }
            }
}

// ---------------------------------------------------------------------------
// Candidate projection + GRU update + residual output
// ---------------------------------------------------------------------------
__global__ __launch_bounds__(256) void k_proj_cand(
    const bf16_t* __restrict__ Xg, const bf16_t* __restrict__ WcT,
    const float* __restrict__ bc, const float* __restrict__ Rbuf,
    float* __restrict__ Hbuf, const float* __restrict__ x,
    float* __restrict__ out, int t)
{
    const int b = blockIdx.z;
    v8f acc[2][2];
    gemm_core<4, 2, 2, 2>(Xg + (long)b * NN * KC, KC, WcT, KC, KC,
                          blockIdx.y, 0, acc);
    const int lane = threadIdx.x & 31, wave = threadIdx.x >> 5;
    const int wm = wave & 3, wn = wave >> 2, g = lane >> 4, l = lane & 15;
    const long m0 = (long)blockIdx.y * 128;
#pragma unroll
    for (int i = 0; i < 2; ++i)
#pragma unroll
        for (int j = 0; j < 2; ++j)
#pragma unroll
            for (int r = 0; r < 8; ++r) {
                long node = m0 + (wm * 2 + i) * 16 + g * 8 + r;
                int  o    = (wn * 2 + j) * 16 + l;          // 0..63
                float hc  = tanhf(acc[i][j][r] + bc[o]);
                long idx  = ((long)b * NN + node) * DD + o;
                float rr  = Rbuf[idx], h = Hbuf[idx];
                float hn  = rr * h + (1.f - rr) * hc;
                Hbuf[idx] = hn;
                long xo = (((long)b * TT + t) * NN + node) * (long)DD + o;
                out[xo] = x[xo] + hn;
            }
}

// ---------------------------------------------------------------------------
// Packing kernels
// ---------------------------------------------------------------------------
__global__ __launch_bounds__(256) void k_pack_adj(
    const float* __restrict__ adj, bf16_t* __restrict__ Abf,
    bf16_t* __restrict__ AbfT)
{
    long i = (long)blockIdx.x * 256 + threadIdx.x;   // NN*NN
    int k = (int)(i & (NN - 1));
    int m = (int)(i >> 11);
    bf16_t v = (bf16_t)adj[i];
    Abf[i] = v;
    AbfT[(long)k * NN + m] = v;                      // transpose for B operand
}

__global__ __launch_bounds__(256) void k_pack_w(
    const float* __restrict__ Wg, const float* __restrict__ Wc,
    bf16_t* __restrict__ WgT, bf16_t* __restrict__ WcT)
{
    int i = blockIdx.x * 256 + threadIdx.x;
    if (i < 128 * KC) {                              // WgT[o][k*128+c]
        int kc = i % KC, o = i / KC, k = kc >> 7, c = kc & 127;
        WgT[i] = (bf16_t)Wg[(k * 128 + c) * 128 + o];
    }
    if (i < DD * KC) {                               // WcT[o][k*128+c]
        int kc = i % KC, o = i / KC, k = kc >> 7, c = kc & 127;
        WcT[i] = (bf16_t)Wc[(k * 128 + c) * DD + o];
    }
}

// inp_t = concat(x_t, h): write transposed pack (B operand) + Xg slab 0
__global__ __launch_bounds__(256) void k_pack_step(
    const float* __restrict__ x, const float* __restrict__ Hbuf,
    bf16_t* __restrict__ XpT, bf16_t* __restrict__ Xg, int t)
{
    long i = (long)blockIdx.x * 256 + threadIdx.x;   // B*NN*128
    int c = (int)(i & 127);
    long rest = i >> 7;
    int node = (int)(rest & (NN - 1));
    int b = (int)(rest >> 11);
    float v = (c < DD)
        ? x[(((long)b * TT + t) * NN + node) * DD + c]
        : Hbuf[((long)b * NN + node) * DD + (c - DD)];
    bf16_t bv = (bf16_t)v;
    XpT[((long)(b * 128 + c)) * NN + node] = bv;
    Xg[((long)b * NN + node) * KC + c]     = bv;
}

// ---------------------------------------------------------------------------
// Launcher
// ---------------------------------------------------------------------------
extern "C" void kernel_launch(void* const* d_in, const int* in_sizes, int n_in,
                              void* d_out, int out_size, void* d_ws, size_t ws_size,
                              hipStream_t stream)
{
    const float* x   = (const float*)d_in[0];   // [B,T,N,D]
    const float* h0  = (const float*)d_in[1];   // [B,N,D]
    const float* adj = (const float*)d_in[2];   // [N,N]
    const float* Wg  = (const float*)d_in[3];   // [3,128,128]
    const float* bg  = (const float*)d_in[4];   // [128]
    const float* Wc  = (const float*)d_in[5];   // [3,128,64]
    const float* bc  = (const float*)d_in[6];   // [64]
    float* out = (float*)d_out;                 // [B,T,N,D] ++ [B,N,D]

    char* w = (char*)d_ws;
    bf16_t* Abf  = (bf16_t*)w; w += (long)NN * NN * 2;            // 8 MB
    bf16_t* AbfT = (bf16_t*)w; w += (long)NN * NN * 2;            // 8 MB
    bf16_t* A2bf = (bf16_t*)w; w += (long)NN * NN * 2;            // 8 MB
    bf16_t* XpT  = (bf16_t*)w; w += (long)BB * 128 * NN * 2;      // 4 MB
    bf16_t* Xg   = (bf16_t*)w; w += (long)BB * NN * KC * 2;       // 12 MB
    bf16_t* WgT  = (bf16_t*)w; w += (long)128 * KC * 2;
    bf16_t* WcT  = (bf16_t*)w; w += (long)DD * KC * 2;
    float*  Rbuf = (float*)w;  w += (long)BB * NN * DD * 4;       // 4 MB
    float*  Hbuf = (float*)w;  w += (long)BB * NN * DD * 4;       // 4 MB

    // --- precompute: bf16 supports + transposed weights ---
    k_pack_adj<<<(NN * NN) / 256, 256, 0, stream>>>(adj, Abf, AbfT);
    k_pack_w<<<(128 * KC + 255) / 256, 256, 0, stream>>>(Wg, Wc, WgT, WcT);
    hipMemcpyAsync(Hbuf, h0, (long)BB * NN * DD * 4,
                   hipMemcpyDeviceToDevice, stream);
    k_gemm_A2<<<dim3(NN / 128, NN / 128, 1), 256, 0, stream>>>(Abf, AbfT, A2bf);

    // --- recurrent scan over T ---
    for (int t = 0; t < TT; ++t) {
        k_pack_step<<<(BB * NN * 128) / 256, 256, 0, stream>>>(x, Hbuf, XpT, Xg, t);
        // gate gconv spatial: slabs 1,2 of Xg   (z picks A vs A2)
        k_spatial<<<dim3((BB * 128) / 128, NN / 128, 2), 256, 0, stream>>>(
            Abf, A2bf, XpT, Xg);
        // gate projection + sigmoid; repack z*h into XpT/Xg slab 0
        k_proj_gate<<<dim3(1, NN / 128, BB), 256, 0, stream>>>(
            Xg, WgT, bg, Hbuf, XpT, Rbuf);
        // candidate gconv spatial on updated pack
        k_spatial<<<dim3((BB * 128) / 128, NN / 128, 2), 256, 0, stream>>>(
            Abf, A2bf, XpT, Xg);
        // candidate projection + tanh + GRU update + residual out
        k_proj_cand<<<dim3(1, NN / 128, BB), 256, 0, stream>>>(
            Xg, WcT, bc, Rbuf, Hbuf, x, out, t);
    }

    // last_state output
    hipMemcpyAsync(out + (long)BB * TT * NN * DD, Hbuf,
                   (long)BB * NN * DD * 4, hipMemcpyDeviceToDevice, stream);
}
